// Net_60103772340724
// MI455X (gfx1250) — matrix operand order repaired and checked by reading, not compiled, not run on hardware
//
#include <hip/hip_runtime.h>
#include <math.h>

typedef __attribute__((ext_vector_type(16))) __bf16 v16bf;
typedef __attribute__((ext_vector_type(8)))  float  v8f;
typedef __attribute__((ext_vector_type(4)))  unsigned int u32x4;

namespace {
constexpr int kN    = 100000;   // nodes
constexpr int kE    = 200000;   // edges
constexpr int kMem  = 172;      // memory/feature dim
constexpr int kTD   = 100;      // time-enc dim
constexpr int kK    = 10;       // neighbors
constexpr int kMsg  = 616;      // 2*MEM + FEAT + TD
constexpr int kMsgP = 640;      // padded K (mult of 32)
constexpr int kG3   = 516;      // 3*MEM
constexpr int kMemP = 192;      // 172 -> pad 192
constexpr int kQinP = 288;      // 272 -> pad 288
constexpr int kCatP = 352;      // 344 -> pad 352
constexpr int kNTBig   = 40;    // 516 cols -> 33 tiles -> pad to 40 (NSUB=8)
constexpr int kNTSmall = 12;    // 172 cols -> 11 tiles -> pad to 12 (NSUB=4)
constexpr int kBatch   = 4;     // nodes per wave in fused attention
}

static __device__ __forceinline__ unsigned short f32_to_bf16(float f) {
  unsigned int u = __float_as_uint(f);
  unsigned int r = u + 0x7FFFu + ((u >> 16) & 1u);   // round to nearest even
  return (unsigned short)(r >> 16);
}
static __device__ __forceinline__ float sigmoidf_(float x) {
  return 1.0f / (1.0f + __expf(-x));
}

union Frag { v16bf v; u32x4 u[2]; };

// ---------------------------------------------------------------- last-msg scatter
__global__ void init_lastpos(int* lastpos) {
  int i = blockIdx.x * blockDim.x + threadIdx.x;
  if (i < kN) lastpos[i] = -1;
}
__global__ void scatter_lastpos(const int* __restrict__ src, const int* __restrict__ dst,
                                int* lastpos) {
  int i = blockIdx.x * blockDim.x + threadIdx.x;
  if (i >= 2 * kE) return;
  int node = (i < kE) ? src[i] : dst[i - kE];
  atomicMax(&lastpos[node], i);
}

// ---------------------------------------------------------------- q bias: tenc(0) @ Wq[172:,:]
__global__ void qbias_kernel(const float* __restrict__ Wq, const float* __restrict__ time_b,
                             float* __restrict__ qbias) {
  int j = blockIdx.x * blockDim.x + threadIdx.x;
  if (j >= kMem) return;
  float s = 0.f;
  for (int k = 0; k < kTD; ++k)
    s += cosf(time_b[k]) * Wq[(size_t)(kMem + k) * kMem + j];
  qbias[j] = s;
}

// ---------------------------------------------------------------- pack B into WMMA fragment layout
// frag layout (16-bit B, 16x16x32): lane = (j%16) + 16*((k%16)>=8), half = (k%8) + 8*((k%32)>=16)
// storage: out[(((kt*NT)+nt)*32 + lane)*16 + half]; pad tiles zero-filled
__global__ void pack_b(const float* __restrict__ W, unsigned short* __restrict__ out,
                       int Kreal, int Nreal, int Ksteps, int NT, int ldw, int transpose) {
  int idx = blockIdx.x * blockDim.x + threadIdx.x;
  int total = Ksteps * NT * 512;
  if (idx >= total) return;
  int h    = idx & 15;
  int lane = (idx >> 4) & 31;
  int tile = idx >> 9;
  int nt = tile % NT, kt = tile / NT;
  int k = kt * 32 + ((lane & 16) ? 8 : 0) + (h & 7) + ((h & 8) ? 16 : 0);
  int j = nt * 16 + (lane & 15);
  float v = 0.f;
  if (k < Kreal && j < Nreal)
    v = transpose ? W[(size_t)j * ldw + k] : W[(size_t)k * ldw + j];
  out[idx] = f32_to_bf16(v);
}

// ---------------------------------------------------------------- message matrix [N, 640] bf16
__global__ void build_msg(const float* __restrict__ memory, const float* __restrict__ edge_features,
                          const float* __restrict__ timestamps, const float* __restrict__ last_update,
                          const float* __restrict__ time_w, const float* __restrict__ time_b,
                          const int* __restrict__ src, const int* __restrict__ dst,
                          const int* __restrict__ lastpos, unsigned short* __restrict__ M) {
  int idx = blockIdx.x * blockDim.x + threadIdx.x;
  int n = idx / kMsgP, c = idx % kMsgP;
  if (n >= kN) return;
  int p = lastpos[n];
  int sp = p < 0 ? 0 : p;
  int e, a, b;
  if (sp < kE) { e = sp;       a = src[e]; b = dst[e]; }
  else         { e = sp - kE;  a = dst[e]; b = src[e]; }
  float v = 0.f;
  if      (c < kMem)       v = memory[(size_t)a * kMem + c];
  else if (c < 2 * kMem)   v = memory[(size_t)b * kMem + (c - kMem)];
  else if (c < kMsg - kTD) v = edge_features[(size_t)e * kMem + (c - 2 * kMem)];
  else if (c < kMsg) {
    int j = c - (kMsg - kTD);
    float dt = timestamps[e] - last_update[a];
    v = cosf(dt * time_w[j] + time_b[j]);
  }
  M[(size_t)n * kMsgP + c] = f32_to_bf16(v);
}

// ---------------------------------------------------------------- memory -> bf16 [N, 192]
__global__ void conv_mem(const float* __restrict__ memory, unsigned short* __restrict__ out) {
  int idx = blockIdx.x * blockDim.x + threadIdx.x;
  int n = idx / kMemP, j = idx % kMemP;
  if (n >= kN) return;
  out[(size_t)n * kMemP + j] = (j < kMem) ? f32_to_bf16(memory[(size_t)n * kMem + j])
                                          : (unsigned short)0;
}

// ---------------------------------------------------------------- generic bf16 WMMA GEMM
// C[M x Ncols] f32 = A(bf16 row-major, lda halves, K zero-padded) @ Bpacked.
// NT is a multiple of NSUB (pad tiles are zero) -> branch-free unrolled inner loops.
// Wave mapping is n-group-major so co-resident waves share B fragments (WGP$-hot).
template <int KSTEPS, int NSUB>
__global__ __launch_bounds__(256)
void gemm_bf16_wmma(const unsigned short* __restrict__ A, int lda,
                    const unsigned short* __restrict__ Bp,
                    float* __restrict__ C, int ldc,
                    int M, int Ncols, int NT) {
  const int lane = threadIdx.x & 31;
  const int wave = blockIdx.x * (blockDim.x >> 5) + (threadIdx.x >> 5);
  const int nGroups = NT / NSUB;
  const int mTiles = M >> 4;
  if (wave >= mTiles * nGroups) return;
  const int mt  = wave % mTiles;   // consecutive waves: same ng, different mt
  const int ng  = wave / mTiles;
  const int nt0 = ng * NSUB;

  const int row  = (mt << 4) + (lane & 15);
  const int koff = (lane & 16) ? 8 : 0;
  const unsigned short* aptr  = A + (size_t)row * lda + koff;
  const unsigned short* bbase = Bp + ((size_t)nt0 * 32 + lane) * 16;

  v8f acc[NSUB] = {};
#pragma unroll
  for (int kt = 0; kt < KSTEPS; ++kt) {
    Frag af;
    const unsigned short* ap = aptr + kt * 32;
    af.u[0] = *(const u32x4*)(ap);
    af.u[1] = *(const u32x4*)(ap + 16);
    const unsigned short* bp = bbase + (size_t)kt * NT * 512;
#pragma unroll
    for (int t = 0; t < NSUB; ++t) {
      Frag bfr;
      bfr.u[0] = *(const u32x4*)(bp + t * 512);
      bfr.u[1] = *(const u32x4*)(bp + t * 512 + 8);
      acc[t] = __builtin_amdgcn_wmma_f32_16x16x32_bf16(false, af.v, false, bfr.v,
                                                       (short)0, acc[t], false, false);
    }
  }
  const int r0 = (mt << 4) + ((lane & 16) ? 8 : 0);
  const int cb = lane & 15;
#pragma unroll
  for (int t = 0; t < NSUB; ++t) {
    int col = ((nt0 + t) << 4) + cb;
    if (col >= Ncols) continue;
#pragma unroll
    for (int i = 0; i < 8; ++i)
      C[(size_t)(r0 + i) * ldc + col] = acc[t][i];
  }
}

// ---------------------------------------------------------------- GRU elementwise + h (bf16 into A2)
__global__ void gru_kernel(const float* __restrict__ gi, const float* __restrict__ gh,
                           const float* __restrict__ bi, const float* __restrict__ bh,
                           const float* __restrict__ memory, const float* __restrict__ node_features,
                           const int* __restrict__ lastpos, unsigned short* __restrict__ A2) {
  int idx = blockIdx.x * blockDim.x + threadIdx.x;
  int n = idx / kMemP, j = idx % kMemP;
  if (n >= kN) return;
  if (j < kMem) {
    size_t b = (size_t)n * kG3;
    float rg = sigmoidf_(gi[b + j] + bi[j] + gh[b + j] + bh[j]);
    float zg = sigmoidf_(gi[b + kMem + j] + bi[kMem + j] + gh[b + kMem + j] + bh[kMem + j]);
    float ng = tanhf(gi[b + 2 * kMem + j] + bi[2 * kMem + j] +
                     rg * (gh[b + 2 * kMem + j] + bh[2 * kMem + j]));
    float mem  = memory[(size_t)n * kMem + j];
    float hnew = (1.f - zg) * ng + zg * mem;
    float memu = (lastpos[n] >= 0) ? hnew : mem;
    float h = memu + node_features[(size_t)n * kMem + j];
    A2[(size_t)n * kCatP + j] = f32_to_bf16(h);
  } else if (j < kMem + 8) {          // zero concat pad cols 344..351
    A2[(size_t)n * kCatP + 2 * kMem + (j - kMem)] = 0;
  }
}

// ---------------------------------------------------------------- fused 2-head K=10 attention
// one wave per block, 4 nodes per wave: four k_in tiles staged in LDS, Wk/Wv
// fragments hoisted to registers per nt-column and reused across the 4 node-tiles
// (cuts packed-weight traffic 4x). A streams from LDS (ds_load_b128). All syncs
// wave-local (s_wait_dscnt + wave_barrier); EXEC is full around every WMMA.
__global__ __launch_bounds__(32)
void attn_kernel(unsigned short* __restrict__ A2,            // [N,352]: cols 0..171 h, writes 172..343 agg
                 const float* __restrict__ q,                // [N,172]
                 const float* __restrict__ qbias,            // [172]
                 const unsigned short* __restrict__ WkP,     // packed, NT=12
                 const unsigned short* __restrict__ WvP,
                 const int* __restrict__ neighbor_idx,       // [N,10]
                 const float* __restrict__ neighbor_ts,      // [N,10]
                 const float* __restrict__ time_w, const float* __restrict__ time_b,
                 const float* __restrict__ timestamps) {
  __shared__ unsigned short Atile[kBatch][16 * kQinP];       // 36.9 KB
  __shared__ float sc[kBatch][2][16];
  __shared__ float aw[kBatch][2][16];

  const int lane  = threadIdx.x;
  const int node0 = blockIdx.x * kBatch;
  if (node0 >= kN) return;                                   // uniform; kN % kBatch == 0
  const float tnow  = timestamps[kE - 1];
  const float scale = 1.0f / sqrtf(86.0f);

  const int arow = lane & 15;
  const int koff = (lane & 16) ? 8 : 0;
  const int ro   = (lane & 16) ? 8 : 0;

  // ---- build 4 k_in tiles (rows 0..9 real, rest zero) ----
  for (int idx = lane; idx < kBatch * 16 * kQinP; idx += 32) {
    int t   = idx / (16 * kQinP);
    int rem = idx % (16 * kQinP);
    int r = rem / kQinP, c = rem % kQinP;
    int node = node0 + t;
    unsigned short val = 0;
    if (r < kK) {
      if (c < kMem) {
        int nb = neighbor_idx[node * kK + r];
        val = A2[(size_t)nb * kCatP + c];
      } else if (c < kMem + kTD) {
        int j = c - kMem;
        float dt = tnow - neighbor_ts[node * kK + r];
        val = f32_to_bf16(cosf(dt * time_w[j] + time_b[j]));
      }
    }
    Atile[t][rem] = val;
  }
  for (int idx = lane; idx < kBatch * 32; idx += 32) ((float*)sc)[idx] = 0.f;
  asm volatile("s_wait_dscnt 0x0" ::: "memory");
  __builtin_amdgcn_wave_barrier();

  // ---- K pass: per nt-column, hoist 9 Wk fragments, sweep the 4 node-tiles ----
  for (int nt = 0; nt < 11; ++nt) {
    Frag breg[9];
#pragma unroll
    for (int kt = 0; kt < 9; ++kt) {
      const unsigned short* bp = WkP + (((size_t)kt * kNTSmall + nt) * 32 + lane) * 16;
      breg[kt].u[0] = *(const u32x4*)(bp);
      breg[kt].u[1] = *(const u32x4*)(bp + 8);
    }
    int c = nt * 16 + arow;
#pragma unroll
    for (int t = 0; t < kBatch; ++t) {
      v8f acc = {};
#pragma unroll
      for (int kt = 0; kt < 9; ++kt) {
        Frag af;
        const unsigned short* ap = &Atile[t][arow * kQinP + kt * 32 + koff];
        af.u[0] = *(const u32x4*)(ap);
        af.u[1] = *(const u32x4*)(ap + 16);
        acc = __builtin_amdgcn_wmma_f32_16x16x32_bf16(false, af.v, false, breg[kt].v,
                                                      (short)0, acc, false, false);
      }
      if (c < kMem) {
        float qv = (q[(size_t)(node0 + t) * kMem + c] + qbias[c]) * scale;
        int hh = c / 86;
#pragma unroll
        for (int i = 0; i < 8; ++i) {
          int s = ro + i;
          if (s < kK) atomicAdd(&sc[t][hh][s], qv * acc[i]);
        }
      }
    }
  }
  asm volatile("s_wait_dscnt 0x0" ::: "memory");
  __builtin_amdgcn_wave_barrier();

  // ---- softmax: lane < 8 handles (node t = lane>>1, head hh = lane&1) ----
  if (lane < 8) {
    int t = lane >> 1, hh = lane & 1;
    float m = -1e30f;
    for (int s = 0; s < kK; ++s) m = fmaxf(m, sc[t][hh][s]);
    float sum = 0.f;
    for (int s = 0; s < kK; ++s) sum += __expf(sc[t][hh][s] - m);
    for (int s = 0; s < kK; ++s) aw[t][hh][s] = __expf(sc[t][hh][s] - m) / sum;
    for (int s = kK; s < 16; ++s) aw[t][hh][s] = 0.f;
  }
  asm volatile("s_wait_dscnt 0x0" ::: "memory");
  __builtin_amdgcn_wave_barrier();

  // ---- V pass: hoist 9 Wv fragments per nt, combine with attention weights ----
  for (int nt = 0; nt < 11; ++nt) {
    Frag breg[9];
#pragma unroll
    for (int kt = 0; kt < 9; ++kt) {
      const unsigned short* bp = WvP + (((size_t)kt * kNTSmall + nt) * 32 + lane) * 16;
      breg[kt].u[0] = *(const u32x4*)(bp);
      breg[kt].u[1] = *(const u32x4*)(bp + 8);
    }
    int c = nt * 16 + arow;
#pragma unroll
    for (int t = 0; t < kBatch; ++t) {
      v8f acc = {};
#pragma unroll
      for (int kt = 0; kt < 9; ++kt) {
        Frag af;
        const unsigned short* ap = &Atile[t][arow * kQinP + kt * 32 + koff];
        af.u[0] = *(const u32x4*)(ap);
        af.u[1] = *(const u32x4*)(ap + 16);
        acc = __builtin_amdgcn_wmma_f32_16x16x32_bf16(false, af.v, false, breg[kt].v,
                                                      (short)0, acc, false, false);
      }
      float partial = 0.f;
      if (c < kMem) {
        int hh = c / 86;
#pragma unroll
        for (int i = 0; i < 8; ++i) partial += aw[t][hh][ro + i] * acc[i];
      }
      float other = __shfl_xor(partial, 16, 32);
      if (c < kMem && lane < 16)
        A2[(size_t)(node0 + t) * kCatP + kMem + c] = f32_to_bf16(partial + other);
    }
  }
}

// ---------------------------------------------------------------- relu(pre1+b1) -> bf16 [N,192]
__global__ void relu_kernel(const float* __restrict__ pre1, const float* __restrict__ b1,
                            unsigned short* __restrict__ A3) {
  int idx = blockIdx.x * blockDim.x + threadIdx.x;
  int n = idx / kMemP, j = idx % kMemP;
  if (n >= kN) return;
  unsigned short v = 0;
  if (j < kMem) v = f32_to_bf16(fmaxf(pre1[(size_t)n * kMem + j] + b1[j], 0.0f));
  A3[(size_t)n * kMemP + j] = v;
}

// ---------------------------------------------------------------- two-pass mean + head
__global__ void reduce1(const float* __restrict__ emb, float* __restrict__ partials) {
  int b = blockIdx.x;          // 0..255
  int t = threadIdx.x;
  if (t >= kMem) return;
  float s = 0.f;
  for (int n = 1 + b; n < kN; n += 256) s += emb[(size_t)n * kMem + t];
  partials[b * kMem + t] = s;
}
__global__ void reduce2(const float* __restrict__ partials, const float* __restrict__ b2,
                        const float* __restrict__ fc2_w, const float* __restrict__ fc2_b,
                        float* __restrict__ out) {
  __shared__ float f[kMem];
  int t = threadIdx.x;
  if (t < kMem) {
    float s = 0.f;
    for (int b = 0; b < 256; ++b) s += partials[b * kMem + t];
    f[t] = tanhf(s / (float)(kN - 1) + b2[t]);
  }
  __syncthreads();
  if (t == 0) {
    float l0 = fc2_b[0], l1 = fc2_b[1];
    for (int c = 0; c < kMem; ++c) { l0 += f[c] * fc2_w[c * 2]; l1 += f[c] * fc2_w[c * 2 + 1]; }
    float m = fmaxf(l0, l1);
    float e0 = __expf(l0 - m), e1 = __expf(l1 - m);
    out[0] = e0 / (e0 + e1);
    out[1] = e1 / (e0 + e1);
  }
}

// ================================================================ host
extern "C" void kernel_launch(void* const* d_in, const int* in_sizes, int n_in,
                              void* d_out, int out_size, void* d_ws, size_t ws_size,
                              hipStream_t stream) {
  const float* node_features = (const float*)d_in[0];
  const float* memory        = (const float*)d_in[1];
  const float* last_update   = (const float*)d_in[2];
  const float* timestamps    = (const float*)d_in[3];
  const float* edge_features = (const float*)d_in[4];
  const float* neighbor_ts   = (const float*)d_in[5];
  const float* time_w        = (const float*)d_in[6];
  const float* time_b        = (const float*)d_in[7];
  const float* gru_Wi        = (const float*)d_in[8];
  const float* gru_Wh        = (const float*)d_in[9];
  const float* gru_bi        = (const float*)d_in[10];
  const float* gru_bh        = (const float*)d_in[11];
  const float* Wq            = (const float*)d_in[12];
  const float* Wk            = (const float*)d_in[13];
  const float* Wv            = (const float*)d_in[14];
  const float* W1            = (const float*)d_in[15];
  const float* b1            = (const float*)d_in[16];
  const float* W2            = (const float*)d_in[17];
  const float* b2            = (const float*)d_in[18];
  const float* fc2_w         = (const float*)d_in[19];
  const float* fc2_b         = (const float*)d_in[20];
  const int*   sources       = (const int*)d_in[21];
  const int*   destinations  = (const int*)d_in[22];
  const int*   neighbor_idx  = (const int*)d_in[23];
  float* out = (float*)d_out;

  char* ws = (char*)d_ws;
  size_t off = 0;
  auto take = [&](size_t bytes) {
    size_t o = off;
    off += (bytes + 255) & ~(size_t)255;
    return o;
  };
  int*            lastpos = (int*)           (ws + take((size_t)kN * 4));
  float*          qbias   = (float*)         (ws + take(kMem * 4));
  unsigned short* WiP     = (unsigned short*)(ws + take((size_t)20 * kNTBig   * 1024));
  unsigned short* WhP     = (unsigned short*)(ws + take((size_t)6  * kNTBig   * 1024));
  unsigned short* WqP     = (unsigned short*)(ws + take((size_t)6  * kNTSmall * 1024));
  unsigned short* WkP     = (unsigned short*)(ws + take((size_t)9  * kNTSmall * 1024));
  unsigned short* WvP     = (unsigned short*)(ws + take((size_t)9  * kNTSmall * 1024));
  unsigned short* W1P     = (unsigned short*)(ws + take((size_t)11 * kNTSmall * 1024));
  unsigned short* W2P     = (unsigned short*)(ws + take((size_t)6  * kNTSmall * 1024));
  unsigned short* Mmsg    = (unsigned short*)(ws + take((size_t)kN * kMsgP * 2));
  unsigned short* membf   = (unsigned short*)(ws + take((size_t)kN * kMemP * 2));
  float*          gi      = (float*)         (ws + take((size_t)kN * kG3 * 4));
  float*          gh      = (float*)         (ws + take((size_t)kN * kG3 * 4));
  unsigned short* A2      = (unsigned short*)(ws + take((size_t)kN * kCatP * 2));
  float*          qbuf    = (float*)         (ws + take((size_t)kN * kMem * 4));
  float*          pre1    = (float*)         (ws + take((size_t)kN * kMem * 4));
  unsigned short* A3      = (unsigned short*)(ws + take((size_t)kN * kMemP * 2));
  float*          emb     = (float*)         (ws + take((size_t)kN * kMem * 4));
  float*          parts   = (float*)         (ws + take((size_t)256 * kMem * 4));
  (void)ws_size; (void)in_sizes; (void)n_in; (void)out_size;

  auto cdiv = [](long a, long b) { return (int)((a + b - 1) / b); };

  // 1) last message position per node
  init_lastpos<<<cdiv(kN, 256), 256, 0, stream>>>(lastpos);
  scatter_lastpos<<<cdiv(2 * kE, 256), 256, 0, stream>>>(sources, destinations, lastpos);

  // 2) constants + weight packing (bf16 WMMA fragments, NT padded per NSUB)
  qbias_kernel<<<1, 192, 0, stream>>>(Wq, time_b, qbias);
  auto pack = [&](const float* W, unsigned short* o, int Kr, int Nr, int Ks, int NT, int ldw, int tr) {
    int total = Ks * NT * 512;
    pack_b<<<cdiv(total, 256), 256, 0, stream>>>(W, o, Kr, Nr, Ks, NT, ldw, tr);
  };
  pack(gru_Wi, WiP, kMsg, kG3, 20, kNTBig, kMsg, 1);    // B = Wi^T
  pack(gru_Wh, WhP, kMem, kG3, 6, kNTBig, kMem, 1);     // B = Wh^T
  pack(Wq,     WqP, kMem, kMem, 6, kNTSmall, kMem, 0);  // top 172 rows only
  pack(Wk,     WkP, 272,  kMem, 9, kNTSmall, kMem, 0);
  pack(Wv,     WvP, 272,  kMem, 9, kNTSmall, kMem, 0);
  pack(W1,     W1P, 344,  kMem, 11, kNTSmall, kMem, 0);
  pack(W2,     W2P, kMem, kMem, 6, kNTSmall, kMem, 0);

  // 3) activations to bf16
  build_msg<<<cdiv((long)kN * kMsgP, 256), 256, 0, stream>>>(
      memory, edge_features, timestamps, last_update, time_w, time_b,
      sources, destinations, lastpos, Mmsg);
  conv_mem<<<cdiv((long)kN * kMemP, 256), 256, 0, stream>>>(memory, membf);

  // 4) GRU gates: gi = msg @ Wi^T ; gh = mem @ Wh^T   (16x128 tiles, NSUB=8)
  {
    int jobs = (kN / 16) * (kNTBig / 8);
    gemm_bf16_wmma<20, 8><<<cdiv(jobs, 8), 256, 0, stream>>>(Mmsg, kMsgP, WiP, gi, kG3, kN, kG3, kNTBig);
    gemm_bf16_wmma<6, 8><<<cdiv(jobs, 8), 256, 0, stream>>>(membf, kMemP, WhP, gh, kG3, kN, kG3, kNTBig);
  }
  gru_kernel<<<cdiv((long)kN * kMemP, 256), 256, 0, stream>>>(
      gi, gh, gru_bi, gru_bh, memory, node_features, lastpos, A2);

  // 5) q = h @ Wq_top  (time-enc part folded into qbias)
  {
    int jobs = (kN / 16) * (kNTSmall / 4);
    gemm_bf16_wmma<6, 4><<<cdiv(jobs, 8), 256, 0, stream>>>(A2, kCatP, WqP, qbuf, kMem, kN, kMem, kNTSmall);
  }

  // 6) fused temporal attention -> agg written into A2 cols 172..343
  attn_kernel<<<kN / kBatch, 32, 0, stream>>>(A2, qbuf, qbias, WkP, WvP,
                                              neighbor_idx, neighbor_ts, time_w, time_b, timestamps);

  // 7) MLP: relu([h|agg] @ W1 + b1) @ W2   (b2 folded into reduction)
  {
    int jobs = (kN / 16) * (kNTSmall / 4);
    gemm_bf16_wmma<11, 4><<<cdiv(jobs, 8), 256, 0, stream>>>(A2, kCatP, W1P, pre1, kMem, kN, kMem, kNTSmall);
    relu_kernel<<<cdiv((long)kN * kMemP, 256), 256, 0, stream>>>(pre1, b1, A3);
    gemm_bf16_wmma<6, 4><<<cdiv(jobs, 8), 256, 0, stream>>>(A3, kMemP, W2P, emb, kMem, kN, kMem, kNTSmall);
  }

  // 8) mean over nodes 1..N-1, tanh, fc2, softmax
  reduce1<<<256, 192, 0, stream>>>(emb, parts);
  reduce2<<<1, 192, 0, stream>>>(parts, b2, fc2_w, fc2_b, out);
}
